// URAClusteringGNN_59631325938205
// MI455X (gfx1250) — matrix operand order repaired and checked by reading, not compiled, not run on hardware
//
#include <hip/hip_runtime.h>
#include <cstdint>

#define NEG_SLOPE 0.2f
#define BN_EPS    1e-5f
#define NN   50000
#define NE   800000
#define CHN  64
#define EDIM 3
#define KCLS 4

typedef __attribute__((ext_vector_type(16))) __bf16 vbf16x16;
typedef __attribute__((ext_vector_type(8)))  float  v8f;

__device__ __forceinline__ unsigned short f2bf(float f) {
  unsigned int u = __float_as_uint(f);
  unsigned int r = (u + 0x7FFFu + ((u >> 16) & 1u)) >> 16;
  return (unsigned short)r;
}

__device__ __forceinline__ void atomicMaxF(float* a, float v) {
  if (v >= 0.f) atomicMax((int*)a, __float_as_int(v));
  else          atomicMin((unsigned int*)a, __float_as_uint(v));
}

// ---------------- generic fill ----------------
__global__ void k_fill(float* p, long long n, float v) {
  long long i = (long long)blockIdx.x * blockDim.x + threadIdx.x;
  if (i < n) p[i] = v;
}

// ---------------- input projection: h = x@Wp + bp ----------------
__global__ void k_lin0(const float* x, const float* Wp, const float* bp, float* h) {
  long long i = (long long)blockIdx.x * blockDim.x + threadIdx.x;
  if (i >= (long long)NN * CHN) return;
  int n = (int)(i >> 6), c = (int)(i & 63);
  h[i] = x[2 * n] * Wp[c] + x[2 * n + 1] * Wp[CHN + c] + bp[c];
}

// ---------------- degree + self-loop edge_attr (scatter mean) ----------------
__global__ void k_deg_acc(const int* dst, const float* ea, float* deg, float* eas) {
  int e = blockIdx.x * blockDim.x + threadIdx.x;
  if (e >= NE) return;
  int d = dst[e];
  atomicAdd(&deg[d], 1.f);
  atomicAdd(&eas[(long long)d * 3 + 0], ea[(long long)e * 3 + 0]);
  atomicAdd(&eas[(long long)d * 3 + 1], ea[(long long)e * 3 + 1]);
  atomicAdd(&eas[(long long)d * 3 + 2], ea[(long long)e * 3 + 2]);
}
__global__ void k_ealoop(float* eas, const float* deg) {
  int n = blockIdx.x * blockDim.x + threadIdx.x;
  if (n >= NN) return;
  float dg = fmaxf(deg[n], 1.f);
  eas[(long long)n * 3 + 0] /= dg;
  eas[(long long)n * 3 + 1] /= dg;
  eas[(long long)n * 3 + 2] /= dg;
}

// ---------------- f32 -> bf16 conversion ----------------
__global__ void k_cvt(const float* s, unsigned short* d, long long n) {
  long long i = (long long)blockIdx.x * blockDim.x + threadIdx.x;
  if (i < n) d[i] = f2bf(s[i]);
}
// transpose W (K x M, row-major) -> Wt (M x K, row-major, bf16)
__global__ void k_wt(const float* W, unsigned short* Wt, int K, int M) {
  int i = blockIdx.x * blockDim.x + threadIdx.x;
  if (i >= K * M) return;
  int m = i / K, k = i % K;
  Wt[i] = f2bf(W[(long long)k * M + m]);
}

// ---------------- WMMA GEMM: C(N x M) = A(N x K, bf16) * Wt(M x K, bf16)^T --------
// One wave computes a 16x16 output tile; K stepped by 32 with v_wmma_f32_16x16x32_bf16.
// A operand layout (ISA 16-bit A 16x32): lane = M row (lane&15);
//   VGPR i holds K = (lane>=16 ? 8 : 0) + (i<4 ? {2i,2i+1} : 16+{2(i-4),...}).
// B operand layout (ISA 16-bit B 32x16): lane = N col (lane&15);
//   VGPR i holds K = (lane>=16 ? 8 : 0) + (i<4 ? {2i,2i+1} : 16+{2(i-4),...}).
__global__ __launch_bounds__(256) void k_gemm(const unsigned short* A,
                                              const unsigned short* Bt,
                                              float* C, int K, int M) {
  const int lane = threadIdx.x & 31;
  const int wave = threadIdx.x >> 5;
  const int rowTile = blockIdx.x * 8 + wave;      // 3125 row tiles (N = 50000)
  const int colTile = blockIdx.y;
  if (rowTile >= (NN >> 4)) return;
  const int hlf = lane >> 4;                      // half-lane select
  const int l16 = lane & 15;
  const unsigned short* ap = A  + (long long)(rowTile * 16 + l16) * K;
  const unsigned short* bp = Bt + (long long)(colTile * 16 + l16) * K;
  v8f acc = {0.f, 0.f, 0.f, 0.f, 0.f, 0.f, 0.f, 0.f};
  for (int k0 = 0; k0 < K; k0 += 32) {
    union { vbf16x16 v; unsigned int u[8]; } a, b;
    const int kba = k0 + hlf * 8;
    const int kbb = k0 + hlf * 8;
#pragma unroll
    for (int i = 0; i < 8; ++i) {
      int koffA = kba + ((i & 4) ? 16 : 0) + 2 * (i & 3);
      int koffB = kbb + ((i & 4) ? 16 : 0) + 2 * (i & 3);
      a.u[i] = *(const unsigned int*)(ap + koffA);
      b.u[i] = *(const unsigned int*)(bp + koffB);
    }
    acc = __builtin_amdgcn_wmma_f32_16x16x32_bf16(false, a.v, false, b.v,
                                                  (short)0, acc, false, false);
  }
  // D layout: VGPR r -> row r (lane<16) / row r+8 (lane>=16), col = lane&15
  float* cp = C + (long long)(rowTile * 16 + hlf * 8) * M + colTile * 16 + l16;
#pragma unroll
  for (int r = 0; r < 8; ++r) cp[(long long)r * M] = acc[r];
}

// ---------------- attention coefficients a_s, a_d ----------------
__global__ void k_coef(const float* hlin, const float* as_, const float* ad_,
                       float* aS, float* aD, int H) {
  long long i = (long long)blockIdx.x * blockDim.x + threadIdx.x;
  if (i >= (long long)NN * H) return;
  int n = (int)(i / H), h = (int)(i % H);
  const float* hp = hlin + (long long)n * H * CHN + h * CHN;
  const float* sp = as_ + h * CHN;
  const float* dp = ad_ + h * CHN;
  float ss = 0.f, dd = 0.f;
#pragma unroll 8
  for (int c = 0; c < CHN; ++c) { float v = hp[c]; ss += v * sp[c]; dd += v * dp[c]; }
  aS[i] = ss; aD[i] = dd;
}

// ---------------- fold We, a_edge into v[EDIM][H] ----------------
__global__ void k_ev(const float* We, const float* ae, float* v, int H) {
  int i = threadIdx.x;
  if (i >= EDIM * H) return;
  int d = i / H, h = i % H;
  float s = 0.f;
  for (int c = 0; c < CHN; ++c) s += We[(long long)d * H * CHN + h * CHN + c] * ae[h * CHN + c];
  v[i] = s;
}

// ---------------- edge pass 1: logits + segment max ----------------
__global__ void k_logit(const int* src, const int* dst, const float* ea, const float* eal,
                        const float* v, const float* aS, const float* aD,
                        float* att, float* mx, int H) {
  long long i = (long long)blockIdx.x * blockDim.x + threadIdx.x;
  long long tot = (long long)(NE + NN) * H;
  if (i >= tot) return;
  int e = (int)(i / H), h = (int)(i % H);
  int s, d; const float* er;
  if (e < NE) { s = src[e]; d = dst[e]; er = ea + (long long)e * 3; }
  else        { s = d = e - NE;         er = eal + (long long)(e - NE) * 3; }
  float aE = er[0] * v[0 * H + h] + er[1] * v[1 * H + h] + er[2] * v[2 * H + h];
  float l = aS[(long long)s * H + h] + aD[(long long)d * H + h] + aE;
  l = l > 0.f ? l : NEG_SLOPE * l;
  att[i] = l;
  atomicMaxF(&mx[(long long)d * H + h], l);
}

// ---------------- edge pass 2: exp + segment sum ----------------
__global__ void k_exp(const int* dst, float* att, const float* mx, float* den, int H) {
  long long i = (long long)blockIdx.x * blockDim.x + threadIdx.x;
  long long tot = (long long)(NE + NN) * H;
  if (i >= tot) return;
  int e = (int)(i / H), h = (int)(i % H);
  int d = (e < NE) ? dst[e] : (e - NE);
  float ex = __expf(att[i] - mx[(long long)d * H + h]);
  att[i] = ex;
  atomicAdd(&den[(long long)d * H + h], ex);
}

// ---------------- edge pass 3: weighted scatter of h[src] ----------------
__global__ void k_scatter(const int* src, const int* dst, const float* att, const float* den,
                          const float* hlin, float* out, int H) {
  long long i = (long long)blockIdx.x * blockDim.x + threadIdx.x;
  long long tot = (long long)(NE + NN) * H;
  if (i >= tot) return;
  int e = (int)(i / H), h = (int)(i % H);
  int s, d;
  if (e < NE) { s = src[e]; d = dst[e]; } else { s = d = e - NE; }
  float alpha = att[i] / (den[(long long)d * H + h] + 1e-16f);
  const float* hp = hlin + (long long)s * H * CHN + h * CHN;
  float*       op = out  + (long long)d * H * CHN + h * CHN;
#pragma unroll 4
  for (int c = 0; c < CHN; ++c) atomicAdd(&op[c], alpha * hp[c]);
}

// ---------------- bias add ----------------
__global__ void k_bias(float* h, const float* b, int C) {
  long long i = (long long)blockIdx.x * blockDim.x + threadIdx.x;
  if (i >= (long long)NN * C) return;
  h[i] += b[(int)(i % C)];
}

// ---------------- BatchNorm stats (partial per-block, atomic combine) -------
__global__ void k_bnstats(const float* h, float* st, int C) {
  int t = threadIdx.x;
  if (t >= C) return;
  int r0 = blockIdx.x * 128;
  int r1 = min(r0 + 128, NN);
  float s = 0.f, s2 = 0.f;
  for (int r = r0; r < r1; ++r) { float x = h[(long long)r * C + t]; s += x; s2 += x * x; }
  atomicAdd(&st[t], s);
  atomicAdd(&st[C + t], s2);
}
__global__ void k_bnelu(float* h, const float* st, int C) {
  long long i = (long long)blockIdx.x * blockDim.x + threadIdx.x;
  if (i >= (long long)NN * C) return;
  int c = (int)(i % C);
  float mu  = st[c] * (1.f / NN);
  float var = st[C + c] * (1.f / NN) - mu * mu;
  float y = (h[i] - mu) * rsqrtf(var + BN_EPS);
  h[i] = y > 0.f ? y : (__expf(y) - 1.f);
}
__global__ void k_elu(float* h, long long n) {
  long long i = (long long)blockIdx.x * blockDim.x + threadIdx.x;
  if (i >= n) return;
  float y = h[i];
  h[i] = y > 0.f ? y : (__expf(y) - 1.f);
}

// ---------------- head: logits + softmax ----------------
__global__ void k_head(const float* h, const float* Wc, const float* bc, float* out) {
  int n = blockIdx.x * blockDim.x + threadIdx.x;
  if (n >= NN) return;
  float l[KCLS];
#pragma unroll
  for (int k = 0; k < KCLS; ++k) {
    float s = bc[k];
    for (int c = 0; c < CHN; ++c) s += h[(long long)n * CHN + c] * Wc[c * KCLS + k];
    l[k] = s;
  }
  float mx = l[0];
#pragma unroll
  for (int k = 1; k < KCLS; ++k) mx = fmaxf(mx, l[k]);
  float sum = 0.f;
#pragma unroll
  for (int k = 0; k < KCLS; ++k) { float e = __expf(l[k] - mx); l[k] = e; sum += e; }
  float inv = 1.f / sum;
#pragma unroll
  for (int k = 0; k < KCLS; ++k) out[(long long)n * KCLS + k] = l[k] * inv;
}

extern "C" void kernel_launch(void* const* d_in, const int* in_sizes, int n_in,
                              void* d_out, int out_size, void* d_ws, size_t ws_size,
                              hipStream_t stream) {
  (void)in_sizes; (void)n_in; (void)out_size; (void)ws_size;
  const float* x   = (const float*)d_in[0];
  const int*   ei  = (const int*)  d_in[1];
  const float* ea  = (const float*)d_in[2];
  const float* Wp  = (const float*)d_in[3];
  const float* bp  = (const float*)d_in[4];
  const float* W1  = (const float*)d_in[5];
  const float* as1 = (const float*)d_in[6];
  const float* ad1 = (const float*)d_in[7];
  const float* We1 = (const float*)d_in[8];
  const float* ae1 = (const float*)d_in[9];
  const float* b1  = (const float*)d_in[10];
  const float* W2  = (const float*)d_in[11];
  const float* as2 = (const float*)d_in[12];
  const float* ad2 = (const float*)d_in[13];
  const float* We2 = (const float*)d_in[14];
  const float* ae2 = (const float*)d_in[15];
  const float* b2  = (const float*)d_in[16];
  const float* W3  = (const float*)d_in[17];
  const float* as3 = (const float*)d_in[18];
  const float* ad3 = (const float*)d_in[19];
  const float* We3 = (const float*)d_in[20];
  const float* ae3 = (const float*)d_in[21];
  const float* b3  = (const float*)d_in[22];
  const float* Wc  = (const float*)d_in[23];
  const float* bc  = (const float*)d_in[24];
  const int* srcI = ei;
  const int* dstI = ei + NE;

  // carve workspace
  char* ws = (char*)d_ws; size_t off = 0;
  auto take = [&](size_t bytes) -> char* {
    char* p = ws + off; off += (bytes + 255) & ~(size_t)255; return p;
  };
  float*          A   = (float*)take((size_t)NN * 256 * 4);   // ping: layer input / accum out
  float*          B   = (float*)take((size_t)NN * 256 * 4);   // pong: h_lin (GEMM out)
  unsigned short* hbf = (unsigned short*)take((size_t)NN * 256 * 2);
  unsigned short* Wt  = (unsigned short*)take((size_t)256 * 256 * 2);
  float*          aS  = (float*)take((size_t)NN * 4 * 4);
  float*          aD  = (float*)take((size_t)NN * 4 * 4);
  float*          Mx  = (float*)take((size_t)NN * 4 * 4);
  float*          Den = (float*)take((size_t)NN * 4 * 4);
  float*          Att = (float*)take((size_t)(NE + NN) * 4 * 4);
  float*          Deg = (float*)take((size_t)NN * 4);
  float*          EaL = (float*)take((size_t)NN * 3 * 4);
  float*          Vv  = (float*)take(64);
  float*          St  = (float*)take(512 * 4);

  dim3 b256(256);
  auto g1 = [](long long n) { return dim3((unsigned)((n + 255) / 256)); };

  // degree + self-loop mean edge_attr (shared by all layers)
  k_fill<<<g1(NN), b256, 0, stream>>>(Deg, NN, 0.f);
  k_fill<<<g1((long long)NN * 3), b256, 0, stream>>>(EaL, (long long)NN * 3, 0.f);
  k_deg_acc<<<g1(NE), b256, 0, stream>>>(dstI, ea, Deg, EaL);
  k_ealoop<<<g1(NN), b256, 0, stream>>>(EaL, Deg);

  // input projection + BN + ELU  (C = 64)
  k_lin0<<<g1((long long)NN * CHN), b256, 0, stream>>>(x, Wp, bp, A);
  k_fill<<<g1(128), b256, 0, stream>>>(St, 128, 0.f);
  k_bnstats<<<dim3((NN + 127) / 128), b256, 0, stream>>>(A, St, CHN);
  k_bnelu<<<g1((long long)NN * CHN), b256, 0, stream>>>(A, St, CHN);

  auto gat = [&](const float* W, const float* as_, const float* ad_, const float* We,
                 const float* ae_, const float* bb, int H, int Cin, int Cout, bool bn) {
    long long tot = (long long)(NE + NN) * H;
    k_cvt<<<g1((long long)NN * Cin), b256, 0, stream>>>(A, hbf, (long long)NN * Cin);
    k_wt<<<g1((long long)Cin * Cout), b256, 0, stream>>>(W, Wt, Cin, Cout);
    k_gemm<<<dim3(391, Cout / 16), b256, 0, stream>>>(hbf, Wt, B, Cin, Cout);
    k_coef<<<g1((long long)NN * H), b256, 0, stream>>>(B, as_, ad_, aS, aD, H);
    k_ev<<<1, 32, 0, stream>>>(We, ae_, Vv, H);
    k_fill<<<g1((long long)NN * H), b256, 0, stream>>>(Mx, (long long)NN * H, -INFINITY);
    k_fill<<<g1((long long)NN * H), b256, 0, stream>>>(Den, (long long)NN * H, 0.f);
    k_logit<<<g1(tot), b256, 0, stream>>>(srcI, dstI, ea, EaL, Vv, aS, aD, Att, Mx, H);
    k_exp<<<g1(tot), b256, 0, stream>>>(dstI, Att, Mx, Den, H);
    k_fill<<<g1((long long)NN * Cout), b256, 0, stream>>>(A, (long long)NN * Cout, 0.f);
    k_scatter<<<g1(tot), b256, 0, stream>>>(srcI, dstI, Att, Den, B, A, H);
    k_bias<<<g1((long long)NN * Cout), b256, 0, stream>>>(A, bb, Cout);
    if (bn) {
      k_fill<<<g1(2 * Cout), b256, 0, stream>>>(St, 2 * Cout, 0.f);
      k_bnstats<<<dim3((NN + 127) / 128), b256, 0, stream>>>(A, St, Cout);
      k_bnelu<<<g1((long long)NN * Cout), b256, 0, stream>>>(A, St, Cout);
    } else {
      k_elu<<<g1((long long)NN * Cout), b256, 0, stream>>>(A, (long long)NN * Cout);
    }
  };

  gat(W1, as1, ad1, We1, ae1, b1, 4, 64, 256, true);
  gat(W2, as2, ad2, We2, ae2, b2, 4, 256, 256, true);
  gat(W3, as3, ad3, We3, ae3, b3, 1, 256, 64, false);

  k_head<<<g1(NN), b256, 0, stream>>>(A, Wc, bc, (float*)d_out);
}
